// RANNet_69939247448806
// MI455X (gfx1250) — matrix operand order repaired and checked by reading, not compiled, not run on hardware
//
#include <hip/hip_runtime.h>

#define B_  128
#define S_  2048
#define D_  256
#define H_  256

typedef __attribute__((ext_vector_type(16))) __bf16        v16bf;
typedef __attribute__((ext_vector_type(8)))  float         v8f;
typedef __attribute__((ext_vector_type(8)))  unsigned short u16x8;
typedef int i32x4_ __attribute__((vector_size(16)));                    // async-LDS builtin src type
typedef __attribute__((address_space(3))) i32x4_ lds_i32x4;             // async-LDS builtin dst type

#if __has_builtin(__builtin_amdgcn_global_load_async_to_lds_b128) && \
    __has_builtin(__builtin_amdgcn_s_wait_asynccnt)
#define USE_ASYNC_LDS 1
#endif

static __device__ __forceinline__ unsigned short f2bf(float f) {
  unsigned u = __float_as_uint(f);
  u += 0x7fffu + ((u >> 16) & 1u);           // round-to-nearest-even
  return (unsigned short)(u >> 16);
}
static __device__ __forceinline__ float bf2f(unsigned short h) {
  return __uint_as_float(((unsigned)h) << 16);
}
static __device__ __forceinline__ float sigmoidf_(float x) {
  return 1.0f / (1.0f + __expf(-x));
}

// Copy 16 bf16 (32B) global -> LDS. Async (ASYNCcnt-tracked, no VGPR staging)
// when the toolchain exposes the gfx1250 async builtins; manual otherwise.
static __device__ __forceinline__ void stage32B(const unsigned short* src,
                                                unsigned short* dstLds) {
#ifdef USE_ASYNC_LDS
  __builtin_amdgcn_global_load_async_to_lds_b128(
      (i32x4_*)(void*)(unsigned short*)src,
      (lds_i32x4*)dstLds, 0, 0);
  __builtin_amdgcn_global_load_async_to_lds_b128(
      (i32x4_*)(void*)(unsigned short*)(src + 8),
      (lds_i32x4*)(dstLds + 8), 0, 0);
#else
  *(u16x8*)dstLds       = *(const u16x8*)src;
  *(u16x8*)(dstLds + 8) = *(const u16x8*)(src + 8);
#endif
}
static __device__ __forceinline__ void wait_async_lds() {
#ifdef USE_ASYNC_LDS
  __builtin_amdgcn_s_wait_asynccnt(0);
#endif
}

// A-fragment (16x32 bf16, row-major source): lane m=L%16, elems e -> K=16*hi+e
// => 16 contiguous bf16 (two 16B chunks).
static __device__ __forceinline__ v16bf afrag(const unsigned short* row, int hi) {
  union { v16bf v; u16x8 h[2]; } u;
  u.h[0] = *(const u16x8*)(row + 16 * hi);
  u.h[1] = *(const u16x8*)(row + 16 * hi + 8);
  return u.v;
}
// B-fragment (32x16 bf16): lane n=L%16, elems e -> K=16*hi+e => 16 contiguous bf16.
static __device__ __forceinline__ v16bf bfrag(const unsigned short* p) {
  union { v16bf v; u16x8 h[2]; } u;
  u.h[0] = *(const u16x8*)(p);
  u.h[1] = *(const u16x8*)(p + 8);
  return u.v;
}

// ---------------------------------------------------------------------------
// prep: cast/concat weights to bf16 ([n][k] layout == W row-major), concat biases
// ---------------------------------------------------------------------------
__global__ void prep_kernel(
    const float* Wf0, const float* Wi0, const float* Wx0,
    const float* Wf1, const float* Wi1, const float* Wx1,
    const float* Uf0, const float* Ui0, const float* Uf1, const float* Ui1,
    const float* fcW,
    const float* bf0, const float* bi0, const float* bx0,
    const float* bf1, const float* bi1, const float* bx1,
    unsigned short* wcat0, unsigned short* wcat1,
    unsigned short* U0, unsigned short* U1, unsigned short* fcwb,
    float* bcat0, float* bcat1)
{
  const int idx = blockIdx.x * 256 + threadIdx.x;
  const int NW = 11 * 65536;
  if (idx < NW) {
    const float* s[11] = {Wf0, Wi0, Wx0, Wf1, Wi1, Wx1, Uf0, Ui0, Uf1, Ui1, fcW};
    const int c = idx >> 16, off = idx & 65535;
    unsigned short* d;
    if (c < 3)       d = wcat0 + c * 65536 + off;
    else if (c < 6)  d = wcat1 + (c - 3) * 65536 + off;
    else if (c < 8)  d = U0    + (c - 6) * 65536 + off;
    else if (c < 10) d = U1    + (c - 8) * 65536 + off;
    else             d = fcwb  + off;
    *d = f2bf(s[c][off]);
  } else {
    const int j = idx - NW;
    if (j < 768) {
      const float* sb[3] = {bf0, bi0, bx0};
      bcat0[j] = sb[j >> 8][j & 255];
    } else if (j < 1536) {
      const int k = j - 768;
      const float* sb[3] = {bf1, bi1, bx1};
      bcat1[k] = sb[k >> 8][k & 255];
    }
  }
}

// ---------------------------------------------------------------------------
// Tiled 128x128 WMMA GEMM: C[m,n] = sum_k A[m,k]*Bw[n,k] + bias[n]
// K fixed = 256, double-buffered LDS pipeline: async-stage tile k+1 while
// WMMAs consume tile k. 8 waves, each computes 32(M) x 64(N) (2x4 16x16 tiles).
// ---------------------------------------------------------------------------
template<bool ABF16, bool OUTF32>
__global__ __launch_bounds__(256) void gemm128_wmma(
    const void* __restrict__ Ap, const unsigned short* __restrict__ Bw,
    const float* __restrict__ bias, void* __restrict__ Cp, int ldc)
{
  __shared__ __align__(128) unsigned short As[2][128][32];   // [buf][m][k]
  __shared__ __align__(128) unsigned short Bs[2][128][32];   // [buf][n][k]
  const int tid  = threadIdx.x;
  const int lane = tid & 31, w = tid >> 5;
  const int hi = lane >> 4, lm = lane & 15;
  const int wm = w & 3, wn = w >> 2;
  const long m0 = (long)blockIdx.x * 128;
  const int  n0 = blockIdx.y * 128;
  const int r  = tid >> 1;            // staging row 0..127
  const int hk = (tid & 1) * 16;      // staging k-half

  const v8f vz = {0.f,0.f,0.f,0.f,0.f,0.f,0.f,0.f};
  v8f acc[2][4];
  #pragma unroll
  for (int i = 0; i < 2; i++)
    #pragma unroll
    for (int j = 0; j < 4; j++) acc[i][j] = vz;

  const unsigned short* Abf = (const unsigned short*)Ap;
  const float*          Af  = (const float*)Ap;

  // ---- stage tile for k0 into buffer `buf` ----
  auto stageA = [&](int k0, int buf) {
    if (ABF16) {
      stage32B(Abf + (m0 + r) * 256 + k0 + hk, &As[buf][r][hk]);
    } else {
      const float4* src = (const float4*)(Af + (m0 + r) * 256 + k0 + hk);
      float4 fa[4] = {src[0], src[1], src[2], src[3]};
      float tmp[16];
      #pragma unroll
      for (int q = 0; q < 4; q++) {
        tmp[q*4+0] = fa[q].x; tmp[q*4+1] = fa[q].y;
        tmp[q*4+2] = fa[q].z; tmp[q*4+3] = fa[q].w;
      }
      #pragma unroll
      for (int q = 0; q < 16; q++) As[buf][r][hk + q] = f2bf(tmp[q]);
      if (k0 < 224) __builtin_prefetch((const void*)(src + 8), 0, 0);
    }
  };
  auto stageB = [&](int k0, int buf) {
    stage32B(Bw + (n0 + r) * 256 + k0 + hk, &Bs[buf][r][hk]);
  };

  stageA(0, 0);
  stageB(0, 0);

  #pragma unroll
  for (int i = 0; i < 8; i++) {
    const int buf = i & 1;
    wait_async_lds();      // our async loads for tile i landed in LDS
    __syncthreads();       // everyone's loads landed; compute(i-1) fully drained
    if (i < 7) {           // overlap: stage tile i+1 into the other buffer
      stageA((i + 1) * 32, buf ^ 1);
      stageB((i + 1) * 32, buf ^ 1);
    }
    #pragma unroll
    for (int mt = 0; mt < 2; mt++) {
      const int m = (wm * 2 + mt) * 16 + lm;
      v16bf a = afrag(&As[buf][m][0], hi);
      #pragma unroll
      for (int nt = 0; nt < 4; nt++) {
        const int n = (wn * 4 + nt) * 16 + lm;
        v16bf b = bfrag(&Bs[buf][n][16 * hi]);
        acc[mt][nt] = __builtin_amdgcn_wmma_f32_16x16x32_bf16(
            false, a, false, b, (short)0, acc[mt][nt], false, false);
      }
    }
  }
  // Epilogue: C/D layout m = 8*hi + vgpr, n = lane%16
  #pragma unroll
  for (int mt = 0; mt < 2; mt++)
    #pragma unroll
    for (int nt = 0; nt < 4; nt++) {
      const int n = n0 + (wn * 4 + nt) * 16 + lm;
      const float bv = bias[n];
      #pragma unroll
      for (int v = 0; v < 8; v++) {
        const long m = m0 + (wm * 2 + mt) * 16 + hi * 8 + v;
        const float val = acc[mt][nt][v] + bv;
        if (OUTF32) ((float*)Cp)[m * ldc + n] = val;
        else        ((unsigned short*)Cp)[m * ldc + n] = f2bf(val);
      }
    }
}

// ---------------------------------------------------------------------------
// Recurrence: 8 workgroups x 16 batch rows. Per step: z(16x512) = hbf @ Ucat^T
// (8 waves x 4 n-tiles x 8 k-steps of WMMA), then fused sigmoid/blend update.
// LAYER==0 streams h out as bf16; LAYER==1 stores only the final state (fp32).
// ---------------------------------------------------------------------------
template<int LAYER>
__global__ __launch_bounds__(256) void rannet_rec(
    const unsigned short* __restrict__ g,   // [B*S][768] bf16 gates (xf|xi|xp)
    const unsigned short* __restrict__ U,   // [512][256] bf16 (Uf|Ui)
    unsigned short* __restrict__ hout,      // LAYER==0: [B*S][256] bf16
    float* __restrict__ h2last)             // LAYER==1: [B][256] fp32
{
  __shared__ __align__(128) unsigned short hbf[16][256];  // bf16 mirror (WMMA A)
  __shared__ __align__(128) float          hfp[16][256];  // fp32 master state
  __shared__ __align__(128) float          zbuf[16][512]; // z exchange
  const int tid  = threadIdx.x;
  const int lane = tid & 31, w = tid >> 5;
  const int hi = lane >> 4, lm = lane & 15;
  const int b0 = blockIdx.x * 16;
  for (int i = tid; i < 16 * 256; i += 256) { hfp[i >> 8][i & 255] = 0.f; hbf[i >> 8][i & 255] = 0; }
  __syncthreads();
  const int um = tid >> 4;            // update row 0..15
  const int nb = (tid & 15) * 16;     // update col base
  const v8f vz = {0.f,0.f,0.f,0.f,0.f,0.f,0.f,0.f};

  for (int t = 0; t < S_; t++) {
    v8f z[4];
    #pragma unroll
    for (int nt = 0; nt < 4; nt++) z[nt] = vz;
    #pragma unroll
    for (int ks = 0; ks < 8; ks++) {
      const int kb = ks * 32;
      v16bf a = afrag(&hbf[lm][kb], hi);
      #pragma unroll
      for (int nt = 0; nt < 4; nt++) {
        const int n = w * 64 + nt * 16 + lm;
        v16bf b = bfrag(&U[n * 256 + kb + 16 * hi]);   // L2-resident
        z[nt] = __builtin_amdgcn_wmma_f32_16x16x32_bf16(
            false, a, false, b, (short)0, z[nt], false, false);
      }
    }
    #pragma unroll
    for (int nt = 0; nt < 4; nt++)
      #pragma unroll
      for (int v = 0; v < 8; v++)
        zbuf[hi * 8 + v][w * 64 + nt * 16 + lm] = z[nt][v];
    __syncthreads();

    const long row = (long)(b0 + um) * S_ + t;
    const unsigned short* grow = g + row * 768;
    u16x8 gf0 = *(const u16x8*)(grow + nb),       gf1 = *(const u16x8*)(grow + nb + 8);
    u16x8 gi0 = *(const u16x8*)(grow + 256 + nb), gi1 = *(const u16x8*)(grow + 256 + nb + 8);
    u16x8 gp0 = *(const u16x8*)(grow + 512 + nb), gp1 = *(const u16x8*)(grow + 512 + nb + 8);
    if (t < S_ - 1) {  // prefetch next timestep's gate row behind the WMMA phase
      __builtin_prefetch((const void*)(grow + 768 + nb), 0, 0);
      __builtin_prefetch((const void*)(grow + 768 + 256 + nb), 0, 0);
      __builtin_prefetch((const void*)(grow + 768 + 512 + nb), 0, 0);
    }
    unsigned short hb[16];
    float hv[16];
    #pragma unroll
    for (int j = 0; j < 16; j++) {
      const float gf = bf2f(j < 8 ? gf0[j] : gf1[j - 8]);
      const float gi = bf2f(j < 8 ? gi0[j] : gi1[j - 8]);
      const float gp = bf2f(j < 8 ? gp0[j] : gp1[j - 8]);
      const float f  = sigmoidf_(gf + zbuf[um][nb + j]);
      const float ii = sigmoidf_(gi + zbuf[um][nb + j + 256]);
      const float hn = f * hfp[um][nb + j] + ii * gp;
      hv[j] = hn; hb[j] = f2bf(hn);
      hfp[um][nb + j] = hn;
      hbf[um][nb + j] = hb[j];
    }
    if (LAYER == 0) {
      u16x8 o0, o1;
      #pragma unroll
      for (int j = 0; j < 8; j++) { o0[j] = hb[j]; o1[j] = hb[j + 8]; }
      *(u16x8*)(hout + row * H_ + nb)     = o0;
      *(u16x8*)(hout + row * H_ + nb + 8) = o1;
    } else if (t == S_ - 1) {
      #pragma unroll
      for (int j = 0; j < 16; j++) h2last[(b0 + um) * H_ + nb + j] = hv[j];
    }
    __syncthreads();
  }
}

// ---------------------------------------------------------------------------
extern "C" void kernel_launch(void* const* d_in, const int* in_sizes, int n_in,
                              void* d_out, int out_size, void* d_ws, size_t ws_size,
                              hipStream_t stream) {
  (void)in_sizes; (void)n_in; (void)out_size; (void)ws_size;
  const float* x   = (const float*)d_in[0];
  const float* Wf0 = (const float*)d_in[1];
  const float* bf0 = (const float*)d_in[2];
  const float* Uf0 = (const float*)d_in[3];
  const float* Wi0 = (const float*)d_in[4];
  const float* bi0 = (const float*)d_in[5];
  const float* Ui0 = (const float*)d_in[6];
  const float* Wx0 = (const float*)d_in[7];
  const float* bx0 = (const float*)d_in[8];
  const float* Wf1 = (const float*)d_in[9];
  const float* bf1 = (const float*)d_in[10];
  const float* Uf1 = (const float*)d_in[11];
  const float* Wi1 = (const float*)d_in[12];
  const float* bi1 = (const float*)d_in[13];
  const float* Ui1 = (const float*)d_in[14];
  const float* Wx1 = (const float*)d_in[15];
  const float* bx1 = (const float*)d_in[16];
  const float* fcW = (const float*)d_in[17];
  const float* fcb = (const float*)d_in[18];

  char* ws = (char*)d_ws;
  unsigned short* wcat0 = (unsigned short*)(ws + 0);         // 768x256 bf16
  unsigned short* wcat1 = (unsigned short*)(ws + 393216);    // 768x256 bf16
  unsigned short* U0    = (unsigned short*)(ws + 786432);    // 512x256 bf16
  unsigned short* U1    = (unsigned short*)(ws + 1048576);   // 512x256 bf16
  unsigned short* fcwb  = (unsigned short*)(ws + 1310720);   // 256x256 bf16
  float*          bcat0 = (float*)(ws + 1441792);            // 768 f32
  float*          bcat1 = (float*)(ws + 1444864);            // 768 f32
  float*          h2last= (float*)(ws + 1447936);            // 128x256 f32
  unsigned short* h1    = (unsigned short*)(ws + 1579008);   // BSx256 bf16 (128MB)
  unsigned short* g     = (unsigned short*)(ws + 1579008 + 134217728); // BSx768 bf16 (384MB, reused)

  const int prep_threads = 11 * 65536 + 1536;
  prep_kernel<<<(prep_threads + 255) / 256, 256, 0, stream>>>(
      Wf0, Wi0, Wx0, Wf1, Wi1, Wx1, Uf0, Ui0, Uf1, Ui1, fcW,
      bf0, bi0, bx0, bf1, bi1, bx1,
      wcat0, wcat1, U0, U1, fcwb, bcat0, bcat1);

  // Layer 0 fused gate GEMM: (B*S x 256) x (256 x 768)
  gemm128_wmma<false, false><<<dim3((B_ * S_) / 128, 768 / 128), 256, 0, stream>>>(
      (const void*)x, wcat0, bcat0, (void*)g, 768);
  // Layer 0 recurrence -> h1 (bf16)
  rannet_rec<0><<<B_ / 16, 256, 0, stream>>>(g, U0, h1, nullptr);
  // Layer 1 fused gate GEMM (A in bf16), reuse gate buffer
  gemm128_wmma<true, false><<<dim3((B_ * S_) / 128, 768 / 128), 256, 0, stream>>>(
      (const void*)h1, wcat1, bcat1, (void*)g, 768);
  // Layer 1 recurrence -> only final state
  rannet_rec<1><<<B_ / 16, 256, 0, stream>>>(g, U1, nullptr, h2last);
  // Output head: (128 x 256) x (256 x 256) -> fp32 d_out
  gemm128_wmma<false, true><<<dim3(1, 256 / 128), 256, 0, stream>>>(
      (const void*)h2last, fcwb, fcb, d_out, 256);
}
